// QBottleneck_36043365548379
// MI455X (gfx1250) — matrix operand order repaired and checked by reading, not compiled, not run on hardware
//
#include <hip/hip_runtime.h>
#include <math.h>
#include <float.h>

typedef __attribute__((ext_vector_type(16))) _Float16 v16h;
typedef __attribute__((ext_vector_type(8)))  float    v8f;

// Problem constants
#define K_CODES 1024
#define D_HID   64
#define N_VEC   65536      // B*H*W = 16*64*64
#define HW      4096       // H*W
#define BSTRIDE 262144     // D_HID*HW
#define N_MT    4096       // N_VEC / 16 (M-tiles)

// d_out layout (floats): latent | commit | embed | indices | distances
#define OUT_LOSS1 4194304u
#define OUT_LOSS2 4194305u
#define OUT_IDX   4194306u
#define OUT_DIST  4259842u

// workspace byte offsets
#define OFF_CBN   0u          // 1024*64 f32        = 262144
#define OFF_CBSQ  262144u     // 1024 f32           =   4096
#define OFF_BHI   266240u     // 2*64*32*16 f16     = 131072
#define OFF_BLO   397312u     // f16                = 131072
#define OFF_AHI   528384u     // 4096*2*32*16 f16   = 8388608
#define OFF_ALO   8916992u    // f16                = 8388608
#define OFF_LSQ   17305600u   // 65536 f32          = 262144
#define OFF_IDXW  17567744u   // 65536 i32          = 262144
#define OFF_PART  17829888u   // 16384 f32          =  65536

__device__ __forceinline__ v8f wmma_f16(v16h a, v16h b, v8f c) {
  // v_wmma_f32_16x16x32_f16 (wave32); 8-arg codegen-confirmed signature
  return __builtin_amdgcn_wmma_f32_16x16x32_f16(
      /*neg_a=*/false, a, /*neg_b=*/false, b,
      /*c_mod=*/(short)0, c, /*reuse_a=*/false, /*reuse_b=*/false);
}

// --- 1) normalize codebook rows, per-row ||cb||^2 -------------------------
__global__ void k_cbnorm(const float* __restrict__ cb,
                         float* __restrict__ cbn, float* __restrict__ cbsq) {
  int t = blockIdx.x * 256 + threadIdx.x;
  if (t >= K_CODES) return;
  float s = 0.f;
  for (int d = 0; d < D_HID; ++d) { float v = cb[t * D_HID + d]; s += v * v; }
  float nrm = fmaxf(sqrtf(s), 1e-12f);
  float ss = 0.f;
  for (int d = 0; d < D_HID; ++d) {
    float v = cb[t * D_HID + d] / nrm;      // reference divides, match rounding
    cbn[t * D_HID + d] = v;
    ss += v * v;
  }
  cbsq[t] = ss;
}

// --- 2) pack B fragments (hi/lo split) in WMMA per-lane layout ------------
// B is 32x16 (KxN): lanes 0-15 -> N=lane, elems 0..7 = K 0..7, 8..15 = K 16..23
//                   lanes 16-31 -> N=lane-16, elems 0..7 = K 8..15, 8..15 = K 24..31
__global__ void k_bpack(const float* __restrict__ cbn,
                        _Float16* __restrict__ bHi, _Float16* __restrict__ bLo) {
  int t = blockIdx.x * 256 + threadIdx.x;          // [0, 2*64*32*16)
  int e = t & 15, lane = (t >> 4) & 31, nb = (t >> 9) & 63, kb = (t >> 15) & 1;
  int code = nb * 16 + (lane & 15);
  int K = (lane < 16) ? ((e < 8) ? e : e + 8) : ((e < 8) ? e + 8 : e + 16);
  int d = kb * 32 + K;
  float v = cbn[code * D_HID + d];
  _Float16 hi = (_Float16)v;
  bHi[t] = hi;
  bLo[t] = (_Float16)(v - (float)hi);
}

// --- 3) ||lat||^2 per vector, fp32, coalesced over hw ---------------------
__global__ void k_latsq(const float* __restrict__ preq, float* __restrict__ lsq) {
  int n = blockIdx.x * 256 + threadIdx.x;
  if (n >= N_VEC) return;
  int b = n >> 12, hw = n & 4095;
  const float* p = preq + (size_t)b * BSTRIDE + hw;
  float s = 0.f;
  for (int d = 0; d < D_HID; ++d) { float v = p[d * HW]; s += v * v; }
  lsq[n] = s;
}

// --- 4) pack A fragments (hi/lo split) in WMMA per-lane layout ------------
// A is 16x32 (MxK): lanes 0-15: M=lane,  elems 0..7=K0..7,  8..15=K16..23
//                   lanes 16-31: M=l-16, elems 0..7=K8..15, 8..15=K24..31
__global__ void k_apack(const float* __restrict__ preq,
                        _Float16* __restrict__ aHi, _Float16* __restrict__ aLo) {
  int t = blockIdx.x * 256 + threadIdx.x;          // [0, 4096*2*32*16)
  int e = t & 15, lane = (t >> 4) & 31, kb = (t >> 9) & 1, mt = t >> 10;
  int M = lane & 15;
  int K = (lane < 16) ? ((e < 8) ? e : e + 8) : ((e < 8) ? e + 8 : e + 16);
  int n = mt * 16 + M;
  int d = kb * 32 + K;
  float v = preq[(size_t)(n >> 12) * BSTRIDE + (size_t)d * HW + (n & 4095)];
  _Float16 hi = (_Float16)v;
  aHi[t] = hi;
  aLo[t] = (_Float16)(v - (float)hi);
}

// --- 5) distances GEMM + per-row argmin -----------------------------------
// One wave per 16-row M-tile; loops all 64 code blocks (K_CODES).
// dist = ||lat||^2 - 2*(lat.cb) + ||cb||^2, dot in hi/lo split f16 WMMA.
__global__ void k_gemm(const _Float16* __restrict__ aHip,
                       const _Float16* __restrict__ aLop,
                       const _Float16* __restrict__ bHip,
                       const _Float16* __restrict__ bLop,
                       const float* __restrict__ lsq,
                       const float* __restrict__ cbsq,
                       int* __restrict__ idxWs,
                       float* __restrict__ outIdx,
                       float* __restrict__ outDist) {
  int wave = threadIdx.x >> 5;
  int lane = threadIdx.x & 31;
  int mt = blockIdx.x * 8 + wave;                   // 512 blocks * 8 waves = 4096
  int laneHalf = lane >> 4;
  int ncol = lane & 15;

  const v16h* aH = (const v16h*)aHip;
  const v16h* aL = (const v16h*)aLop;
  const v16h* bH = (const v16h*)bHip;
  const v16h* bL = (const v16h*)bLop;

  v16h aH0 = aH[(mt * 2 + 0) * 32 + lane];
  v16h aL0 = aL[(mt * 2 + 0) * 32 + lane];
  v16h aH1 = aH[(mt * 2 + 1) * 32 + lane];
  v16h aL1 = aL[(mt * 2 + 1) * 32 + lane];

  float ls[8];
#pragma unroll
  for (int r = 0; r < 8; ++r) ls[r] = lsq[mt * 16 + r + 8 * laneHalf];

  float bestd[8];
  int besti[8];
#pragma unroll
  for (int r = 0; r < 8; ++r) { bestd[r] = FLT_MAX; besti[r] = 0; }

  for (int nb = 0; nb < 64; ++nb) {
    v16h bH0 = bH[(0 * 64 + nb) * 32 + lane];
    v16h bL0 = bL[(0 * 64 + nb) * 32 + lane];
    v16h bH1 = bH[(1 * 64 + nb) * 32 + lane];
    v16h bL1 = bL[(1 * 64 + nb) * 32 + lane];

    v8f acc = {};
    acc = wmma_f16(aL0, bH0, acc);   // lo*hi
    acc = wmma_f16(aH0, bL0, acc);   // hi*lo
    acc = wmma_f16(aH0, bH0, acc);   // hi*hi
    acc = wmma_f16(aL1, bH1, acc);
    acc = wmma_f16(aH1, bL1, acc);
    acc = wmma_f16(aH1, bH1, acc);

    int code = nb * 16 + ncol;
    float cq = cbsq[code];
#pragma unroll
    for (int r = 0; r < 8; ++r) {
      float dist = ls[r] - 2.0f * acc[r] + cq;
      int row = mt * 16 + r + 8 * laneHalf;
      outDist[(size_t)row * K_CODES + code] = dist;
      if (dist < bestd[r]) { bestd[r] = dist; besti[r] = code; }
    }
  }

  // argmin across the 16 lanes holding one row (C layout: N striped on lanes)
#pragma unroll
  for (int r = 0; r < 8; ++r) {
    float d = bestd[r];
    int i = besti[r];
#pragma unroll
    for (int m = 8; m >= 1; m >>= 1) {
      float od = __shfl_xor(d, m, 16);
      int oi = __shfl_xor(i, m, 16);
      if (od < d || (od == d && oi < i)) { d = od; i = oi; }
    }
    bestd[r] = d;
    besti[r] = i;
  }
  if ((lane & 15) == 0) {
#pragma unroll
    for (int r = 0; r < 8; ++r) {
      int row = mt * 16 + r + 8 * laneHalf;
      idxWs[row] = besti[r];
      outIdx[row] = (float)besti[r];
    }
  }
}

// --- 6) gather q = cbn[idx], latent = p + (q - p), loss partials ----------
__global__ void k_gather(const float* __restrict__ preq,
                         const float* __restrict__ cbn,
                         const int* __restrict__ idx,
                         float* __restrict__ latent,
                         float* __restrict__ partial) {
  int bi = blockIdx.x;                 // 16*64*16 = 16384 blocks
  int b = bi >> 10;
  int rem = bi & 1023;
  int d = rem >> 4;
  int hw = ((rem & 15) << 8) + threadIdx.x;
  int n = b * HW + hw;
  size_t off = (size_t)b * BSTRIDE + (size_t)d * HW + hw;
  float p = preq[off];
  float q = cbn[idx[n] * D_HID + d];
  float diff = q - p;
  latent[off] = p + diff;              // straight-through, reference rounding
  __shared__ float sdata[256];
  sdata[threadIdx.x] = diff * diff;
  __syncthreads();
  for (int s = 128; s > 0; s >>= 1) {
    if (threadIdx.x < s) sdata[threadIdx.x] += sdata[threadIdx.x + s];
    __syncthreads();
  }
  if (threadIdx.x == 0) partial[bi] = sdata[0];
}

// --- 7) deterministic final reduction -> both loss scalars ----------------
__global__ void k_reduce(const float* __restrict__ partial, float* __restrict__ out) {
  __shared__ float sdata[256];
  float s = 0.f;
  for (int i = threadIdx.x; i < 16384; i += 256) s += partial[i];
  sdata[threadIdx.x] = s;
  __syncthreads();
  for (int st = 128; st > 0; st >>= 1) {
    if (threadIdx.x < st) sdata[threadIdx.x] += sdata[threadIdx.x + st];
    __syncthreads();
  }
  if (threadIdx.x == 0) {
    float m = sdata[0] / 4194304.0f;   // mean over B*D*H*W
    out[OUT_LOSS1] = m;                // commitment
    out[OUT_LOSS2] = m;                // embedding (same value numerically)
  }
}

extern "C" void kernel_launch(void* const* d_in, const int* in_sizes, int n_in,
                              void* d_out, int out_size, void* d_ws, size_t ws_size,
                              hipStream_t stream) {
  (void)in_sizes; (void)n_in; (void)out_size; (void)ws_size;
  const float* preq = (const float*)d_in[0];
  const float* cb   = (const float*)d_in[1];
  float* out = (float*)d_out;
  char* ws = (char*)d_ws;

  float*    cbn  = (float*)(ws + OFF_CBN);
  float*    cbsq = (float*)(ws + OFF_CBSQ);
  _Float16* bHi  = (_Float16*)(ws + OFF_BHI);
  _Float16* bLo  = (_Float16*)(ws + OFF_BLO);
  _Float16* aHi  = (_Float16*)(ws + OFF_AHI);
  _Float16* aLo  = (_Float16*)(ws + OFF_ALO);
  float*    lsq  = (float*)(ws + OFF_LSQ);
  int*      idxW = (int*)(ws + OFF_IDXW);
  float*    part = (float*)(ws + OFF_PART);

  k_cbnorm<<<4, 256, 0, stream>>>(cb, cbn, cbsq);
  k_bpack<<<256, 256, 0, stream>>>(cbn, bHi, bLo);
  k_latsq<<<256, 256, 0, stream>>>(preq, lsq);
  k_apack<<<16384, 256, 0, stream>>>(preq, aHi, aLo);
  k_gemm<<<512, 256, 0, stream>>>(aHi, aLo, bHi, bLo, lsq, cbsq,
                                  idxW, out + OUT_IDX, out + OUT_DIST);
  k_gather<<<16384, 256, 0, stream>>>(preq, cbn, idxW, out, part);
  k_reduce<<<1, 256, 0, stream>>>(part, out);
}
